// TemporalGATConv_12240656794125
// MI455X (gfx1250) — matrix operand order repaired and checked by reading, not compiled, not run on hardware
//
#include <hip/hip_runtime.h>
#include <hip/hip_fp16.h>

#define NN   50000
#define EE   400000
#define QDIM 96      // MEM + TF
#define KDIM 128     // MEM + EF + TF
#define HD   128     // H*D
#define XDIM 192     // HD + MEM
#define HIDD 512

typedef __attribute__((ext_vector_type(16))) _Float16 v16h;
typedef __attribute__((ext_vector_type(8)))  _Float16 v8h;
typedef __attribute__((ext_vector_type(8)))  float    v8f;

// ---- workspace layout (bytes) ----
#define OFF_WQ   0u          // 12288 halves
#define OFF_WK   24576u      // 16384 halves
#define OFF_W1   57344u      // 98304 halves
#define OFF_W2   253952u     // 8192 halves
#define OFF_A    270336u     // E*8 f32
#define OFF_M    13070336u   // N*8 f32
#define OFF_DEN  14670336u   // N*8 f32
#define OFF_AGG  16270336u   // N*128 f32   (end = 41,870,336 B)

// padded LDS row strides (halves) -> conflict-free b128 reads
#define QSTR 104
#define KSTR 136
#define XSTR 200
#define HSTR 520

__device__ __forceinline__ v8f wmma_f16(v16h a, v16h b, v8f c) {
  return __builtin_amdgcn_wmma_f32_16x16x32_f16(false, a, false, b, (short)0, c, false, false);
}

__device__ __forceinline__ v16h frag2(v8h lo, v8h hi) {
  return __builtin_shufflevector(lo, hi, 0,1,2,3,4,5,6,7,8,9,10,11,12,13,14,15);
}

// A fragment (16x32 f16) from an LDS tile, row stride in halves.
__device__ __forceinline__ v16h load_a_lds(const _Float16* tile, int stride, int k0, int lane) {
  const int row = lane & 15;
  const int kb  = (lane >> 4) << 3;      // 0 or 8
  const _Float16* p = tile + row * stride + k0 + kb;
  v8h lo = *(const v8h*)p;               // K = kb .. kb+7
  v8h hi = *(const v8h*)(p + 16);        // K = 16+kb .. 16+kb+7
  return frag2(lo, hi);
}

// B fragment (32x16 f16) from a row-major weight [outdim][rowlen]: column n = weight row n.
__device__ __forceinline__ v16h load_b_glb(const _Float16* w, int rowlen, int n0, int k0, int lane) {
  const int col = lane & 15;
  const int kb  = (lane >> 4) << 3;      // 0 or 8
  const _Float16* p = w + (n0 + col) * rowlen + k0 + kb;
  v8h lo = *(const v8h*)p;
  v8h hi = *(const v8h*)(p + 16);
  return frag2(lo, hi);
}

__device__ __forceinline__ void atomic_max_f32(float* addr, float val) {
  __hip_atomic_fetch_max(addr, val, __ATOMIC_RELAXED, __HIP_MEMORY_SCOPE_AGENT);
}
__device__ __forceinline__ void atomic_add_f32(float* addr, float val) {
  unsafeAtomicAdd(addr, val);
}

// ---------------- P0: weight f16 conversion + accumulator init ----------------
__global__ void tgat_prep(const float* __restrict__ Wq, const float* __restrict__ Wk,
                          const float* __restrict__ W1, const float* __restrict__ W2,
                          _Float16* __restrict__ wq_h, _Float16* __restrict__ wk_h,
                          _Float16* __restrict__ w1_h, _Float16* __restrict__ w2_h,
                          float* __restrict__ m, float* __restrict__ den,
                          float* __restrict__ agg) {
  const int CWQ = HD * QDIM, CWK = HD * KDIM, CW1 = HIDD * XDIM, CW2 = 16 * HIDD;
  const long total = (long)CWQ + CWK + CW1 + CW2 + 2L * NN * 8 + (long)NN * HD;
  for (long i = blockIdx.x * (long)blockDim.x + threadIdx.x; i < total;
       i += (long)gridDim.x * blockDim.x) {
    long j = i;
    if (j < CWQ)    { wq_h[j] = (_Float16)Wq[j]; continue; } j -= CWQ;
    if (j < CWK)    { wk_h[j] = (_Float16)Wk[j]; continue; } j -= CWK;
    if (j < CW1)    { w1_h[j] = (_Float16)W1[j]; continue; } j -= CW1;
    if (j < CW2)    { w2_h[j] = (_Float16)W2[j]; continue; } j -= CW2;
    if (j < NN * 8) { m[j]   = -3.0e38f;         continue; } j -= NN * 8;
    if (j < NN * 8) { den[j] = 0.0f;             continue; } j -= NN * 8;
    agg[j] = 0.0f;
  }
}

// ---------------- P1: attention logits + segment max ----------------
__global__ __launch_bounds__(256)
void tgat_edge_attn(const int* __restrict__ src, const int* __restrict__ dst,
                    const float* __restrict__ memory, const float* __restrict__ ts,
                    const float* __restrict__ efeats, const float* __restrict__ ets,
                    const float* __restrict__ time_w, const float* __restrict__ time_b,
                    const _Float16* __restrict__ wq_h, const _Float16* __restrict__ wk_h,
                    float* __restrict__ a_out, float* __restrict__ m_out) {
  __shared__ _Float16 qs[16 * QSTR];
  __shared__ _Float16 ks[16 * KSTR];
  const int e0 = blockIdx.x << 4;
  const int t  = threadIdx.x;
  {
    const int el = t >> 4, part = t & 15;
    const int e = e0 + el;
    const int sN = src[e], dN = dst[e];
    const float td = ets[e] - ts[sN];
#pragma unroll
    for (int jj = 0; jj < 8; ++jj) {          // key row: mem(64) | efeat(32) | cos(32)
      const int j = part * 8 + jj;
      float v;
      if (j < 64)      v = memory[sN * 64 + j];
      else if (j < 96) v = efeats[e * 32 + (j - 64)];
      else             v = __cosf(td * time_w[j - 96] + time_b[j - 96]);
      ks[el * KSTR + j] = (_Float16)v;
    }
#pragma unroll
    for (int jj = 0; jj < 6; ++jj) {          // query row: mem(64) | cos(time_b)(32)
      const int j = part * 6 + jj;
      float v = (j < 64) ? memory[dN * 64 + j] : __cosf(time_b[j - 64]);
      qs[el * QSTR + j] = (_Float16)v;
    }
  }
  __syncthreads();
  const int lane = t & 31;
  const int h    = t >> 5;                    // one wave per head (D == 16 == one N-tile)
  v8f cq = {}, ck = {};
#pragma unroll
  for (int kt = 0; kt < 3; ++kt)
    cq = wmma_f16(load_a_lds(qs, QSTR, kt * 32, lane),
                  load_b_glb(wq_h, QDIM, h * 16, kt * 32, lane), cq);
#pragma unroll
  for (int kt = 0; kt < 4; ++kt)
    ck = wmma_f16(load_a_lds(ks, KSTR, kt * 32, lane),
                  load_b_glb(wk_h, KDIM, h * 16, kt * 32, lane), ck);
  const int hi = lane >> 4;
  float red[8];
#pragma unroll
  for (int v = 0; v < 8; ++v) {               // a[row][h] = sum_d qh*kh (reduce 16 cols)
    float p = cq[v] * ck[v];
    p += __shfl_xor(p, 1);
    p += __shfl_xor(p, 2);
    p += __shfl_xor(p, 4);
    p += __shfl_xor(p, 8);
    red[v] = p;
  }
  if ((lane & 15) == 0) {
#pragma unroll
    for (int v = 0; v < 8; ++v) {
      const int ee = e0 + v + 8 * hi;
      a_out[ee * 8 + h] = red[v];
      atomic_max_f32(&m_out[dst[ee] * 8 + h], red[v]);
    }
  }
}

// ---------------- P2: exp-weighted scatter aggregation ----------------
__global__ __launch_bounds__(256)
void tgat_edge_aggr(const int* __restrict__ src, const int* __restrict__ dst,
                    const float* __restrict__ memory, const float* __restrict__ ts,
                    const float* __restrict__ efeats, const float* __restrict__ ets,
                    const float* __restrict__ time_w, const float* __restrict__ time_b,
                    const _Float16* __restrict__ wk_h,
                    const float* __restrict__ a_in, const float* __restrict__ m_in,
                    float* __restrict__ den, float* __restrict__ agg) {
  __shared__ _Float16 ks[16 * KSTR];
  const int e0 = blockIdx.x << 4;
  const int t  = threadIdx.x;
  {
    const int el = t >> 4, part = t & 15;
    const int e = e0 + el;
    const int sN = src[e];
    const float td = ets[e] - ts[sN];
#pragma unroll
    for (int jj = 0; jj < 8; ++jj) {
      const int j = part * 8 + jj;
      float v;
      if (j < 64)      v = memory[sN * 64 + j];
      else if (j < 96) v = efeats[e * 32 + (j - 64)];
      else             v = __cosf(td * time_w[j - 96] + time_b[j - 96]);
      ks[el * KSTR + j] = (_Float16)v;
    }
  }
  __syncthreads();
  const int lane = t & 31;
  const int h    = t >> 5;
  v8f ck = {};
#pragma unroll
  for (int kt = 0; kt < 4; ++kt)
    ck = wmma_f16(load_a_lds(ks, KSTR, kt * 32, lane),
                  load_b_glb(wk_h, KDIM, h * 16, kt * 32, lane), ck);
  const int hi = lane >> 4, col = lane & 15;
#pragma unroll
  for (int v = 0; v < 8; ++v) {
    const int ee = e0 + v + 8 * hi;
    const int dN = dst[ee];
    const float w = __expf(a_in[ee * 8 + h] - m_in[dN * 8 + h]);
    atomic_add_f32(&agg[dN * 128 + h * 16 + col], w * ck[v]);
    if (col == 0) atomic_add_f32(&den[dN * 8 + h], w);
  }
}

// ---------------- P3: per-node MLP (normalize -> W1+ReLU -> W2) ----------------
__global__ __launch_bounds__(256)
void tgat_node_mlp(const float* __restrict__ memory, const float* __restrict__ agg,
                   const float* __restrict__ den,
                   const _Float16* __restrict__ w1_h, const float* __restrict__ b1,
                   const _Float16* __restrict__ w2_h, const float* __restrict__ b2,
                   float* __restrict__ out) {
  __shared__ _Float16 xs[16 * XSTR];
  __shared__ _Float16 hs[16 * HSTR];
  __shared__ float    ps[8 * 256];
  const int n0 = blockIdx.x << 4;
  const int t  = threadIdx.x;
  {
    const int nl = t >> 4, part = t & 15;
    const int n = n0 + nl;
#pragma unroll
    for (int jj = 0; jj < 12; ++jj) {         // x = [agg/den/sqrt(D)(128) | memory(64)]
      const int j = part * 12 + jj;
      float v;
      if (j < 128) v = agg[n * 128 + j] / (den[n * 8 + (j >> 4)] * 4.0f);
      else         v = memory[n * 64 + (j - 128)];
      xs[nl * XSTR + j] = (_Float16)v;
    }
  }
  __syncthreads();
  const int lane = t & 31, wv = t >> 5;
  const int hi = lane >> 4, col = lane & 15;
#pragma unroll
  for (int nb = 0; nb < 4; ++nb) {            // GEMM1: each wave owns 4 of 32 N-tiles
    const int nt = wv * 4 + nb;
    v8f c = {};
#pragma unroll
    for (int kt = 0; kt < 6; ++kt)
      c = wmma_f16(load_a_lds(xs, XSTR, kt * 32, lane),
                   load_b_glb(w1_h, XDIM, nt * 16, kt * 32, lane), c);
    const float bb = b1[nt * 16 + col];
#pragma unroll
    for (int v = 0; v < 8; ++v) {
      const float hv = fmaxf(c[v] + bb, 0.0f);
      hs[(v + 8 * hi) * HSTR + nt * 16 + col] = (_Float16)hv;
    }
  }
  __syncthreads();
  v8f c2 = {};
#pragma unroll
  for (int kk = 0; kk < 4; ++kk) {            // GEMM2: split K=512 over 8 waves
    const int kt = wv * 4 + kk;
    c2 = wmma_f16(load_a_lds(hs, HSTR, kt * 32, lane),
                  load_b_glb(w2_h, HIDD, 0, kt * 32, lane), c2);
  }
#pragma unroll
  for (int v = 0; v < 8; ++v) ps[wv * 256 + (v + 8 * hi) * 16 + col] = c2[v];
  __syncthreads();
  {
    const int r = t >> 4, cc = t & 15;
    float s = b2[cc];
#pragma unroll
    for (int w = 0; w < 8; ++w) s += ps[w * 256 + r * 16 + cc];
    out[(n0 + r) * 16 + cc] = s;
  }
}

extern "C" void kernel_launch(void* const* d_in, const int* in_sizes, int n_in,
                              void* d_out, int out_size, void* d_ws, size_t ws_size,
                              hipStream_t stream) {
  const int*   src    = (const int*)  d_in[0];
  const int*   dst    = (const int*)  d_in[1];
  const float* memory = (const float*)d_in[2];
  const float* ts     = (const float*)d_in[3];
  const float* efeats = (const float*)d_in[4];
  const float* ets    = (const float*)d_in[5];
  const float* Wq     = (const float*)d_in[6];
  const float* Wk     = (const float*)d_in[7];
  const float* W1     = (const float*)d_in[8];
  const float* b1     = (const float*)d_in[9];
  const float* W2     = (const float*)d_in[10];
  const float* b2     = (const float*)d_in[11];
  const float* time_w = (const float*)d_in[12];
  const float* time_b = (const float*)d_in[13];
  float* out = (float*)d_out;

  char* ws = (char*)d_ws;
  _Float16* wq_h = (_Float16*)(ws + OFF_WQ);
  _Float16* wk_h = (_Float16*)(ws + OFF_WK);
  _Float16* w1_h = (_Float16*)(ws + OFF_W1);
  _Float16* w2_h = (_Float16*)(ws + OFF_W2);
  float* a_buf = (float*)(ws + OFF_A);
  float* m_buf = (float*)(ws + OFF_M);
  float* den   = (float*)(ws + OFF_DEN);
  float* agg   = (float*)(ws + OFF_AGG);

  const long prep_total = 135168L + 2L * NN * 8 + (long)NN * HD;
  const int prep_blocks = (int)((prep_total + 255) / 256);
  tgat_prep<<<prep_blocks, 256, 0, stream>>>(Wq, Wk, W1, W2, wq_h, wk_h, w1_h, w2_h,
                                             m_buf, den, agg);
  tgat_edge_attn<<<EE / 16, 256, 0, stream>>>(src, dst, memory, ts, efeats, ets,
                                              time_w, time_b, wq_h, wk_h, a_buf, m_buf);
  tgat_edge_aggr<<<EE / 16, 256, 0, stream>>>(src, dst, memory, ts, efeats, ets,
                                              time_w, time_b, wk_h, a_buf, m_buf, den, agg);
  tgat_node_mlp<<<NN / 16, 256, 0, stream>>>(memory, agg, den, w1_h, b1, w2_h, b2, out);
}
